// StackedMoELoRALayer_26431228740181
// MI455X (gfx1250) — matrix operand order repaired and checked by reading, not compiled, not run on hardware
//
#include <hip/hip_runtime.h>

// ---------------------------------------------------------------------------
// StackedMoELoRA on MI455X (gfx1250, wave32, WMMA):
//   out = x @ W^T + b + sum_s g_s * (x @ A_s^T) @ B_s^T
// Reformulated as:
//   Bg[o][s*16+r] = g_s * B[s][o][r]            (tiny build kernel)
//   H = x @ Acat^T            [8192 x 256]      (bf16x3 WMMA GEMM)
//   out = [x|H] @ [W|Bg]^T + b                  (bf16x3 WMMA GEMM, K=4352)
// bf16x3 split (hi*hi + hi*lo + lo*hi, f32 accum) ~ f32 accuracy at 2.7x
// fewer matrix instructions than V_WMMA_F32_16X16X4_F32.
// Software-pipelined: next tile's global_load_b128 issue before the WMMA
// phase so HBM/L2 latency hides under matrix work.
// ---------------------------------------------------------------------------

typedef __attribute__((ext_vector_type(16))) __bf16 v16bf;
typedef __attribute__((ext_vector_type(8)))  __bf16 v8bf;
typedef __attribute__((ext_vector_type(4)))  __bf16 v4bf;
typedef __attribute__((ext_vector_type(8)))  float  v8f;

#define T_TOK   8192
#define D_IN    4096
#define D_OUT   4096
#define RANK    16
#define NSTACK  16
#define KLORA   (NSTACK * RANK)   // 256
#define GATE_TH 0.12f

#define BM 128
#define BN 128
#define BK 32
#define PITCH 40   // bf16 elems per LDS row (32 + 8 pad) -> 80B stride, conflict-free b128 reads

__device__ __forceinline__ void cvt_split4(float4 v, v4bf& hi, v4bf& lo) {
    float f0 = v.x, f1 = v.y, f2 = v.z, f3 = v.w;
    __bf16 h0 = (__bf16)f0, h1 = (__bf16)f1, h2 = (__bf16)f2, h3 = (__bf16)f3;
    hi[0] = h0; hi[1] = h1; hi[2] = h2; hi[3] = h3;
    lo[0] = (__bf16)(f0 - (float)h0);
    lo[1] = (__bf16)(f1 - (float)h1);
    lo[2] = (__bf16)(f2 - (float)h2);
    lo[3] = (__bf16)(f3 - (float)h3);
}

__device__ __forceinline__ v16bf frag_cat(v8bf a, v8bf b) {
    return __builtin_shufflevector(a, b, 0,1,2,3,4,5,6,7,8,9,10,11,12,13,14,15);
}

// Bg[o*256 + s*16 + r] = gate(s) * B[s][o][r]
__global__ __launch_bounds__(256) void build_bg_kernel(
        const float* __restrict__ B, const float* __restrict__ dw,
        const int* __restrict__ sd, float* __restrict__ Bg) {
    int idx = blockIdx.x * 256 + threadIdx.x;       // 0 .. D_OUT*KLORA-1
    int o = idx >> 8;
    int j = idx & 255;
    int s = j >> 4;
    int r = j & 15;
    float w = dw[sd[s]];
    float g = (w > GATE_TH) ? w : 0.0f;
    Bg[idx] = g * B[((size_t)s * D_OUT + o) * RANK + r];
}

// C[M x N] = Acat([A1|A2]) * Bcat([B1|B2])^T (+ bias), f32 in, bf16x3 WMMA
__global__ __launch_bounds__(256) void gemm_bf16x3_kernel(
        const float* __restrict__ A1, const float* __restrict__ A2,
        const float* __restrict__ B1, const float* __restrict__ B2,
        const float* __restrict__ bias, float* __restrict__ C,
        int K1, int K2, int lda1, int lda2, int ldb1, int ldb2, int ldc) {
    __shared__ __bf16 sAhi[BM * PITCH];
    __shared__ __bf16 sAlo[BM * PITCH];
    __shared__ __bf16 sBhi[BN * PITCH];
    __shared__ __bf16 sBlo[BN * PITCH];

    const int tid  = threadIdx.x;
    const int lane = tid & 31;
    const int wid  = tid >> 5;         // 8 waves
    const int wm   = wid & 3;          // wave row: 4 x 32
    const int wn   = wid >> 2;         // wave col: 2 x 64
    const int sel  = lane >> 4;        // half-wave select
    const int lr   = lane & 15;

    const int rowA0 = blockIdx.y * BM;
    const int rowB0 = blockIdx.x * BN;

    const int srow = tid >> 3;         // 0..31 staging row
    const int scol = (tid & 7) * 4;    // 0..28 staging k (float4)

    v8f acc[2][4] = {};

    const int KT = K1 + K2;

    // ---- software pipeline: global tile for step k lives in ra/rb regs ----
    float4 ra[4], rb[4];

    auto gload = [&](int kg) {
        const float* pa; const float* pb; int kl, la, lb;
        if (kg < K1) { pa = A1; pb = B1; kl = kg;      la = lda1; lb = ldb1; }
        else         { pa = A2; pb = B2; kl = kg - K1; la = lda2; lb = ldb2; }
        #pragma unroll
        for (int p = 0; p < 4; ++p) {
            int r = p * 32 + srow;
            ra[p] = *(const float4*)(pa + (size_t)(rowA0 + r) * la + kl + scol);
            rb[p] = *(const float4*)(pb + (size_t)(rowB0 + r) * lb + kl + scol);
        }
    };

    gload(0);  // prologue

    for (int k0 = 0; k0 < KT; k0 += BK) {
        // Convert staged registers (f32 -> bf16 hi/lo) and commit to LDS
        #pragma unroll
        for (int p = 0; p < 4; ++p) {
            int r = p * 32 + srow;
            v4bf h, l;
            cvt_split4(ra[p], h, l);
            *(v4bf*)&sAhi[r * PITCH + scol] = h;
            *(v4bf*)&sAlo[r * PITCH + scol] = l;
            cvt_split4(rb[p], h, l);
            *(v4bf*)&sBhi[r * PITCH + scol] = h;
            *(v4bf*)&sBlo[r * PITCH + scol] = l;
        }
        __syncthreads();

        // Issue next tile's global loads now; they complete under the WMMAs.
        if (k0 + BK < KT) gload(k0 + BK);

        // Load fragments per documented 16-bit WMMA VGPR layouts.
        // A (16x32): half-wave 0 -> K 0-7 then 16-23; half-wave 1 -> K 8-15 then 24-31
        v16bf ahi[2], alo[2], bhi[4], blo[4];
        #pragma unroll
        for (int m = 0; m < 2; ++m) {
            int row = (wm * 32 + m * 16 + lr) * PITCH;
            v8bf c0 = *(const v8bf*)&sAhi[row + sel * 8];
            v8bf c1 = *(const v8bf*)&sAhi[row + 16 + sel * 8];
            ahi[m] = frag_cat(c0, c1);
            c0 = *(const v8bf*)&sAlo[row + sel * 8];
            c1 = *(const v8bf*)&sAlo[row + 16 + sel * 8];
            alo[m] = frag_cat(c0, c1);
        }
        // B (32x16): lane = N (column of C = row of W); half-wave 0 -> K 0-7 then 16-23; half-wave 1 -> K 8-15 then 24-31
        #pragma unroll
        for (int n = 0; n < 4; ++n) {
            int row = (wn * 64 + n * 16 + lr) * PITCH;
            v8bf c0 = *(const v8bf*)&sBhi[row + sel * 8];
            v8bf c1 = *(const v8bf*)&sBhi[row + 16 + sel * 8];
            bhi[n] = frag_cat(c0, c1);
            c0 = *(const v8bf*)&sBlo[row + sel * 8];
            c1 = *(const v8bf*)&sBlo[row + 16 + sel * 8];
            blo[n] = frag_cat(c0, c1);
        }

        // 24 WMMAs per wave per K-step: hi*hi + hi*lo + lo*hi
        #pragma unroll
        for (int m = 0; m < 2; ++m) {
            #pragma unroll
            for (int n = 0; n < 4; ++n) {
                acc[m][n] = __builtin_amdgcn_wmma_f32_16x16x32_bf16(
                    false, ahi[m], false, bhi[n], (short)0, acc[m][n], false, false);
                acc[m][n] = __builtin_amdgcn_wmma_f32_16x16x32_bf16(
                    false, ahi[m], false, blo[n], (short)0, acc[m][n], false, false);
                acc[m][n] = __builtin_amdgcn_wmma_f32_16x16x32_bf16(
                    false, alo[m], false, bhi[n], (short)0, acc[m][n], false, false);
            }
        }
        __syncthreads();
    }

    // Epilogue: C/D layout -> VGPR i: M = i + 8*sel, N = lr (per 16x16 subtile)
    #pragma unroll
    for (int n = 0; n < 4; ++n) {
        int col = rowB0 + wn * 64 + n * 16 + lr;
        float bv = bias ? bias[col] : 0.0f;
        #pragma unroll
        for (int m = 0; m < 2; ++m) {
            int rbase = rowA0 + wm * 32 + m * 16 + sel * 8;
            #pragma unroll
            for (int i = 0; i < 8; ++i) {
                C[(size_t)(rbase + i) * ldc + col] = acc[m][n][i] + bv;
            }
        }
    }
}

extern "C" void kernel_launch(void* const* d_in, const int* in_sizes, int n_in,
                              void* d_out, int out_size, void* d_ws, size_t ws_size,
                              hipStream_t stream) {
    (void)in_sizes; (void)n_in; (void)out_size; (void)ws_size;
    const float* x  = (const float*)d_in[0];   // [T, D_IN]
    const float* W  = (const float*)d_in[1];   // [D_OUT, D_IN]
    const float* b  = (const float*)d_in[2];   // [D_OUT]
    const float* A  = (const float*)d_in[3];   // [S, R, D_IN] == Acat [256][4096]
    const float* Bl = (const float*)d_in[4];   // [S, D_OUT, R]
    const float* dw = (const float*)d_in[5];   // [S]
    const int*   sd = (const int*)d_in[6];     // [S]
    float* out = (float*)d_out;

    float* H  = (float*)d_ws;                                          // [T, 256]  8 MB
    float* Bg = (float*)((char*)d_ws + (size_t)T_TOK * KLORA * 4);     // [D_OUT,256] 4 MB

    // 1) gated, flattened LoRA-B
    build_bg_kernel<<<(D_OUT * KLORA) / 256, 256, 0, stream>>>(Bl, dw, sd, Bg);

    // 2) H = x @ Acat^T  (K = 4096, N = 256)
    gemm_bf16x3_kernel<<<dim3(KLORA / BN, T_TOK / BM), 256, 0, stream>>>(
        x, nullptr, A, nullptr, nullptr, H,
        D_IN, 0, D_IN, 0, D_IN, 0, KLORA);

    // 3) out = [x|H] @ [W|Bg]^T + b  (K = 4096 + 256)
    gemm_bf16x3_kernel<<<dim3(D_OUT / BN, T_TOK / BM), 256, 0, stream>>>(
        x, H, W, Bg, b, out,
        D_IN, KLORA, D_IN, KLORA, D_IN, KLORA, D_OUT);
}
